// caps_att_90864328114677
// MI455X (gfx1250) — matrix operand order repaired and checked by reading, not compiled, not run on hardware
//
#include <hip/hip_runtime.h>
#include <hip/hip_bf16.h>
#include <cstdint>

// Problem constants (match reference)
#define BB 8
#define NN 4096
#define DD 1024
#define OUTD 1024
#define NUM_ITERS 3
#define LN_EPS 1e-5f

typedef __bf16 bf16_t;
typedef __attribute__((ext_vector_type(16))) __bf16 v16bf;
typedef __attribute__((ext_vector_type(8)))  __bf16 v8bf;
typedef __attribute__((ext_vector_type(8)))  float  v8f;

// ---------------------------------------------------------------------------
// WMMA fragment loaders (LDS, bf16). Layout per CDNA5 ISA 7.12.2:
//  A (16x32): lanes 0-15 = rows, K 0..7 & 16..23; lanes 16-31 = rows, K 8..15 & 24..31
//  B (32x16): lane%16 = column, lanes 0-15 hold K 0..7 & 16..23, lanes 16-31 hold K 8..15 & 24..31
// Stride 40 (80B) makes the 16-row b128 fragment loads bank-conflict-free.
// ---------------------------------------------------------------------------
#define LDSSTR 40

__device__ __forceinline__ v16bf load_a_frag(const bf16_t* lds, int rowbase, int lane) {
    int r  = rowbase + (lane & 15);
    int kh = (lane < 16) ? 0 : 8;
    v8bf lo = *(const v8bf*)(lds + r * LDSSTR + kh);
    v8bf hi = *(const v8bf*)(lds + r * LDSSTR + 16 + kh);
    v16bf out;
#pragma unroll
    for (int i = 0; i < 8; ++i) { out[i] = lo[i]; out[i + 8] = hi[i]; }
    return out;
}

__device__ __forceinline__ v16bf load_b_frag(const bf16_t* lds, int colbase, int lane) {
    int c  = colbase + (lane & 15);
    int ko = (lane < 16) ? 0 : 8;
    v8bf lo = *(const v8bf*)(lds + c * LDSSTR + ko);
    v8bf hi = *(const v8bf*)(lds + c * LDSSTR + ko + 16);
    v16bf out;
#pragma unroll
    for (int i = 0; i < 8; ++i) { out[i] = lo[i]; out[i + 8] = hi[i]; }
    return out;
}

// ---------------------------------------------------------------------------
// prep: W_f -> bf16 ; mean over 4 of W_g -> bf16
// ---------------------------------------------------------------------------
__global__ __launch_bounds__(256) void prep_kernel(const float* __restrict__ Wf,
                                                   const float* __restrict__ Wg,
                                                   bf16_t* __restrict__ wf_bf,
                                                   bf16_t* __restrict__ wgm_bf) {
    size_t i  = (size_t)blockIdx.x * 256 + threadIdx.x;
    size_t DS = (size_t)DD * DD;
    wf_bf[i] = (bf16_t)Wf[i];
    float m = 0.25f * (Wg[i] + Wg[i + DS] + Wg[i + 2 * DS] + Wg[i + 3 * DS]);
    wgm_bf[i] = (bf16_t)m;
}

// ---------------------------------------------------------------------------
// b init from mask
// ---------------------------------------------------------------------------
__global__ __launch_bounds__(256) void binit_kernel(const unsigned char* __restrict__ mask,
                                                    float* __restrict__ blogit) {
    int i = blockIdx.x * 256 + threadIdx.x;  // B*N
    blogit[i] = mask[i] ? -1e18f : 0.0f;
}

// ---------------------------------------------------------------------------
// q0 = query @ W_h   (8 x 1024, tiny)
// ---------------------------------------------------------------------------
__global__ __launch_bounds__(128) void qinit_kernel(const float* __restrict__ query,
                                                    const float* __restrict__ Wh,
                                                    float* __restrict__ q) {
    __shared__ float qs[DD];
    int b = blockIdx.y;
    int j = blockIdx.x * 128 + threadIdx.x;
    for (int i = threadIdx.x; i < DD; i += 128) qs[i] = query[b * DD + i];
    __syncthreads();
    float acc = 0.f;
    for (int k = 0; k < DD; ++k) acc = fmaf(qs[k], Wh[(size_t)k * DD + j], acc);
    q[b * DD + j] = acc;
}

// ---------------------------------------------------------------------------
// WcT[j,k] = sum_t Wgm[j,t] * Wf[k,t]   (bf16 in, bf16 out, WMMA)
// Block tile 128x64, 8 waves, wave computes 2x2 of 16x16. Tiny GEMM (2 GFLOP).
// ---------------------------------------------------------------------------
__global__ __launch_bounds__(256) void gemm_wc_kernel(const bf16_t* __restrict__ A,  // Wgm [D,D]
                                                      const bf16_t* __restrict__ Bt, // Wf  [D,D]
                                                      bf16_t* __restrict__ C) {      // WcT [D,D]
    __shared__ bf16_t sA[128 * LDSSTR];
    __shared__ bf16_t sB[64 * LDSSTR];
    int tid = threadIdx.x, lane = tid & 31, wave = tid >> 5;
    int wm = wave >> 1, wn = wave & 1;
    int bm = blockIdx.x * 128, bn = blockIdx.y * 64;
    v8f acc[2][2] = {};
    for (int k0 = 0; k0 < DD; k0 += 32) {
        {   // A tile 128x32 bf16
            int r = tid >> 1, c = (tid & 1) << 4;
            *(v8bf*)(sA + r * LDSSTR + c)     = *(const v8bf*)(A + (size_t)(bm + r) * DD + k0 + c);
            *(v8bf*)(sA + r * LDSSTR + c + 8) = *(const v8bf*)(A + (size_t)(bm + r) * DD + k0 + c + 8);
        }
        {   // B tile 64x32 bf16
            int r = tid >> 2, c = (tid & 3) << 3;
            *(v8bf*)(sB + r * LDSSTR + c) = *(const v8bf*)(Bt + (size_t)(bn + r) * DD + k0 + c);
        }
        __syncthreads();
#pragma unroll
        for (int tm = 0; tm < 2; ++tm) {
            v16bf af = load_a_frag(sA, wm * 32 + tm * 16, lane);
#pragma unroll
            for (int tn = 0; tn < 2; ++tn) {
                v16bf bf = load_b_frag(sB, wn * 32 + tn * 16, lane);
                acc[tm][tn] = __builtin_amdgcn_wmma_f32_16x16x32_bf16(
                    false, af, false, bf, (short)0, acc[tm][tn], false, false);
            }
        }
        __syncthreads();
    }
    int col0 = bn + wn * 32 + (lane & 15);
#pragma unroll
    for (int tm = 0; tm < 2; ++tm) {
        int rbase = bm + wm * 32 + tm * 16 + ((lane < 16) ? 0 : 8);
#pragma unroll
        for (int tn = 0; tn < 2; ++tn)
#pragma unroll
            for (int r = 0; r < 8; ++r)
                C[(size_t)(rbase + r) * DD + col0 + tn * 16] = (bf16_t)acc[tm][tn][r];
    }
}

// ---------------------------------------------------------------------------
// fg[m,j] = sum_k feat[m,k] * WcT[j,k]
// M = B*N = 32768, N = D = 1024, K = D = 1024.
// Block tile 128x128 (8 waves 4Mx2N, wave tile 32x64 = 2x4 WMMA tiles).
// Double-buffered LDS, one barrier per K-step. Grid: x = N-tile (fast) so the
// 8 re-reads of each feat panel stay L2-resident.
// ---------------------------------------------------------------------------
__global__ __launch_bounds__(256) void gemm_fg_kernel(const float* __restrict__ A,   // feat [M,K] fp32
                                                      const bf16_t* __restrict__ Bt, // WcT  [N,K] bf16
                                                      float* __restrict__ C) {       // fg   [M,N] fp32
    __shared__ bf16_t sA[2][128 * LDSSTR];
    __shared__ bf16_t sB[2][128 * LDSSTR];
    int tid = threadIdx.x, lane = tid & 31, wave = tid >> 5;
    int wm = wave >> 1;            // 0..3  -> M offset wm*32
    int wn = wave & 1;             // 0..1  -> N offset wn*64
    int bm = blockIdx.y * 128;     // M panel
    int bn = blockIdx.x * 128;     // N panel (fast-varying for L2 reuse of A)

    // A staging: thread owns 4 float4 (row ra[t], col ca[t])
    int raf[4], caf[4];
#pragma unroll
    for (int t = 0; t < 4; ++t) {
        int idx = tid + t * 256;
        raf[t] = idx >> 3;
        caf[t] = (idx & 7) << 2;
    }
    // B staging: thread owns 2 v8bf at row rb, cols cb, cb+8
    int rb = tid >> 1, cb = (tid & 1) << 4;

    v8f acc[2][4] = {};

    float4 aReg[4];
    v8bf   bReg[2];
    // --- preload K-tile 0 ---
#pragma unroll
    for (int t = 0; t < 4; ++t)
        aReg[t] = *(const float4*)(A + (size_t)(bm + raf[t]) * DD + caf[t]);
    bReg[0] = *(const v8bf*)(Bt + (size_t)(bn + rb) * DD + cb);
    bReg[1] = *(const v8bf*)(Bt + (size_t)(bn + rb) * DD + cb + 8);
#pragma unroll
    for (int t = 0; t < 4; ++t) {
        bf16_t* d = &sA[0][raf[t] * LDSSTR + caf[t]];
        d[0] = (bf16_t)aReg[t].x; d[1] = (bf16_t)aReg[t].y;
        d[2] = (bf16_t)aReg[t].z; d[3] = (bf16_t)aReg[t].w;
    }
    *(v8bf*)&sB[0][rb * LDSSTR + cb]     = bReg[0];
    *(v8bf*)&sB[0][rb * LDSSTR + cb + 8] = bReg[1];

    int cur = 0;
    for (int k0 = 0; k0 < DD; k0 += 32) {
        __syncthreads();  // buf[cur] ready for all waves
        bool more = (k0 + 32) < DD;
        if (more) {   // issue global loads for next tile (latency hidden by WMMAs)
            int kn = k0 + 32;
#pragma unroll
            for (int t = 0; t < 4; ++t) {
                const float* src = A + (size_t)(bm + raf[t]) * DD + kn + caf[t];
                aReg[t] = *(const float4*)src;
            }
            bReg[0] = *(const v8bf*)(Bt + (size_t)(bn + rb) * DD + kn + cb);
            bReg[1] = *(const v8bf*)(Bt + (size_t)(bn + rb) * DD + kn + cb + 8);
        }
        // --- compute on buf[cur]: load 4 B frags once, reuse across both tm ---
        {
            v16bf bf0 = load_b_frag(sB[cur], wn * 64 + 0,  lane);
            v16bf bf1 = load_b_frag(sB[cur], wn * 64 + 16, lane);
            v16bf bf2 = load_b_frag(sB[cur], wn * 64 + 32, lane);
            v16bf bf3 = load_b_frag(sB[cur], wn * 64 + 48, lane);
#pragma unroll
            for (int tm = 0; tm < 2; ++tm) {
                v16bf af = load_a_frag(sA[cur], wm * 32 + tm * 16, lane);
                acc[tm][0] = __builtin_amdgcn_wmma_f32_16x16x32_bf16(
                    false, af, false, bf0, (short)0, acc[tm][0], false, false);
                acc[tm][1] = __builtin_amdgcn_wmma_f32_16x16x32_bf16(
                    false, af, false, bf1, (short)0, acc[tm][1], false, false);
                acc[tm][2] = __builtin_amdgcn_wmma_f32_16x16x32_bf16(
                    false, af, false, bf2, (short)0, acc[tm][2], false, false);
                acc[tm][3] = __builtin_amdgcn_wmma_f32_16x16x32_bf16(
                    false, af, false, bf3, (short)0, acc[tm][3], false, false);
            }
        }
        if (more) {   // stage next tile into the other buffer
            int nxt = cur ^ 1;
#pragma unroll
            for (int t = 0; t < 4; ++t) {
                bf16_t* d = &sA[nxt][raf[t] * LDSSTR + caf[t]];
                d[0] = (bf16_t)aReg[t].x; d[1] = (bf16_t)aReg[t].y;
                d[2] = (bf16_t)aReg[t].z; d[3] = (bf16_t)aReg[t].w;
            }
            *(v8bf*)&sB[nxt][rb * LDSSTR + cb]     = bReg[0];
            *(v8bf*)&sB[nxt][rb * LDSSTR + cb + 8] = bReg[1];
        }
        cur ^= 1;
    }
    // --- store 32x64 wave tile ---
#pragma unroll
    for (int tm = 0; tm < 2; ++tm) {
        int rbase = bm + wm * 32 + tm * 16 + ((lane < 16) ? 0 : 8);
#pragma unroll
        for (int tn = 0; tn < 4; ++tn) {
            int col = bn + wn * 64 + tn * 16 + (lane & 15);
#pragma unroll
            for (int r = 0; r < 8; ++r)
                C[(size_t)(rbase + r) * DD + col] = acc[tm][tn][r];
        }
    }
}

// ---------------------------------------------------------------------------
// reductions helpers
// ---------------------------------------------------------------------------
__device__ __forceinline__ float block_sum(float v, float* red) {
    int lane = threadIdx.x & 31, wave = threadIdx.x >> 5;
#pragma unroll
    for (int o = 16; o > 0; o >>= 1) v += __shfl_down(v, o, 32);
    if (lane == 0) red[wave] = v;
    __syncthreads();
    float r = (threadIdx.x < 8) ? red[threadIdx.x] : 0.f;
    if (wave == 0) {
#pragma unroll
        for (int o = 4; o > 0; o >>= 1) r += __shfl_down(r, o, 32);
        if (lane == 0) red[0] = r;
    }
    __syncthreads();
    float out = red[0];
    __syncthreads();
    return out;
}

__device__ __forceinline__ float block_max(float v, float* red) {
    int lane = threadIdx.x & 31, wave = threadIdx.x >> 5;
#pragma unroll
    for (int o = 16; o > 0; o >>= 1) v = fmaxf(v, __shfl_down(v, o, 32));
    if (lane == 0) red[wave] = v;
    __syncthreads();
    float r = (threadIdx.x < 8) ? red[threadIdx.x] : -3.4e38f;
    if (wave == 0) {
#pragma unroll
        for (int o = 4; o > 0; o >>= 1) r = fmaxf(r, __shfl_down(r, o, 32));
        if (lane == 0) red[0] = r;
    }
    __syncthreads();
    float out = red[0];
    __syncthreads();
    return out;
}

// softmax over N per batch row
__global__ __launch_bounds__(256) void softmax_kernel(const float* __restrict__ blogit,
                                                      float* __restrict__ c) {
    __shared__ float red[8];
    int b = blockIdx.x;
    const float* row = blogit + b * NN;
    float m = -3.4e38f;
    for (int i = threadIdx.x; i < NN; i += 256) m = fmaxf(m, row[i]);
    m = block_max(m, red);
    float s = 0.f;
    for (int i = threadIdx.x; i < NN; i += 256) s += __expf(row[i] - m);
    s = block_sum(s, red);
    float inv = 1.0f / s;
    for (int i = threadIdx.x; i < NN; i += 256) c[b * NN + i] = __expf(row[i] - m) * inv;
}

// delta[b,n] = dot(q[b,:], fg[b,n,:])  — one wave per row
__global__ __launch_bounds__(256) void delta_kernel(const float* __restrict__ q,
                                                    const float* __restrict__ fg,
                                                    float* __restrict__ delta) {
    int gw = blockIdx.x * 8 + (threadIdx.x >> 5);  // 0 .. B*N-1
    int b = gw >> 12, n = gw & (NN - 1);
    int lane = threadIdx.x & 31;
    const float4* qr = (const float4*)(q + b * DD);
    const float4* fr = (const float4*)(fg + ((size_t)b * NN + n) * DD);
    float s = 0.f;
#pragma unroll
    for (int e = 0; e < 8; ++e) {
        float4 a = qr[e * 32 + lane], x = fr[e * 32 + lane];
        s += a.x * x.x + a.y * x.y + a.z * x.z + a.w * x.w;
    }
#pragma unroll
    for (int o = 16; o > 0; o >>= 1) s += __shfl_down(s, o, 32);
    if (lane == 0) delta[gw] = s;
}

// outacc[b,d] += sum over n-chunk of c[b,n]*fg[b,n,d]
__global__ __launch_bounds__(256) void wsum_kernel(const float* __restrict__ c,
                                                   const float* __restrict__ fg,
                                                   float* __restrict__ outacc) {
    int b = blockIdx.y;
    int n0 = blockIdx.x * (NN / 64);
    int tid = threadIdx.x;  // owns float4 #tid
    float4 acc = {0.f, 0.f, 0.f, 0.f};
    const float* crow = c + b * NN;
    for (int n = n0; n < n0 + NN / 64; ++n) {
        float w = crow[n];
        float4 v = *(const float4*)(fg + ((size_t)b * NN + n) * DD + tid * 4);
        acc.x = fmaf(w, v.x, acc.x); acc.y = fmaf(w, v.y, acc.y);
        acc.z = fmaf(w, v.z, acc.z); acc.w = fmaf(w, v.w, acc.w);
    }
    float* o = outacc + b * DD + tid * 4;
    atomicAdd(o + 0, acc.x); atomicAdd(o + 1, acc.y);
    atomicAdd(o + 2, acc.z); atomicAdd(o + 3, acc.w);
}

// b += standardize(delta)  (ddof=1, +1e-9)
__global__ __launch_bounds__(256) void bupdate_kernel(float* __restrict__ blogit,
                                                      const float* __restrict__ delta) {
    __shared__ float red[8];
    int b = blockIdx.x;
    const float* dr = delta + b * NN;
    float s = 0.f, sq = 0.f;
    for (int i = threadIdx.x; i < NN; i += 256) { float v = dr[i]; s += v; sq += v * v; }
    s = block_sum(s, red);
    sq = block_sum(sq, red);
    float mean = s / NN;
    float var = (sq - NN * mean * mean) / (NN - 1);
    float inv = 1.0f / (sqrtf(fmaxf(var, 0.f)) + 1e-9f);
    for (int i = threadIdx.x; i < NN; i += 256)
        blogit[b * NN + i] += (dr[i] - mean) * inv;
}

// q = LayerNorm(q + selu(out * scale))
__global__ __launch_bounds__(256) void qupdate_kernel(float* __restrict__ q,
                                                      const float* __restrict__ outacc,
                                                      const float* __restrict__ lnw,
                                                      const float* __restrict__ lnb,
                                                      float scale) {
    __shared__ float red[8];
    const float SELU_A = 1.6732632423543772f, SELU_S = 1.0507009873554805f;
    int b = blockIdx.x, tid = threadIdx.x;
    float h[4];
    float s = 0.f;
#pragma unroll
    for (int j = 0; j < 4; ++j) {
        int d = tid * 4 + j;
        float o = outacc[b * DD + d] * scale;
        float se = SELU_S * (o > 0.f ? o : SELU_A * (__expf(o) - 1.0f));
        h[j] = q[b * DD + d] + se;
        s += h[j];
    }
    s = block_sum(s, red);
    float mu = s / DD;
    float vq = 0.f;
#pragma unroll
    for (int j = 0; j < 4; ++j) { float dv = h[j] - mu; vq += dv * dv; }
    vq = block_sum(vq, red);
    float inv = rsqrtf(vq / DD + LN_EPS);
#pragma unroll
    for (int j = 0; j < 4; ++j) {
        int d = tid * 4 + j;
        q[b * DD + d] = (h[j] - mu) * inv * lnw[d] + lnb[d];
    }
}

// result[b,o] = dot(q[b,:], W_out[o,:]) + b_out[o]   — wave per output
__global__ __launch_bounds__(256) void final_kernel(const float* __restrict__ q,
                                                    const float* __restrict__ Wout,
                                                    const float* __restrict__ bout,
                                                    float* __restrict__ out) {
    int gw = blockIdx.x * 8 + (threadIdx.x >> 5);  // 0 .. B*OUT-1
    int b = gw >> 10, o = gw & (OUTD - 1);
    int lane = threadIdx.x & 31;
    const float4* qr = (const float4*)(q + b * DD);
    const float4* wr = (const float4*)(Wout + (size_t)o * DD);
    float s = 0.f;
#pragma unroll
    for (int e = 0; e < 8; ++e) {
        float4 a = qr[e * 32 + lane], x = wr[e * 32 + lane];
        s += a.x * x.x + a.y * x.y + a.z * x.z + a.w * x.w;
    }
#pragma unroll
    for (int of = 16; of > 0; of >>= 1) s += __shfl_down(s, of, 32);
    if (lane == 0) out[gw] = s + bout[o];
}

// ---------------------------------------------------------------------------
extern "C" void kernel_launch(void* const* d_in, const int* in_sizes, int n_in,
                              void* d_out, int out_size, void* d_ws, size_t ws_size,
                              hipStream_t stream) {
    const float* query = (const float*)d_in[0];
    const float* feat  = (const float*)d_in[1];
    const float* W_h   = (const float*)d_in[2];
    const float* W_f   = (const float*)d_in[3];
    const float* W_g   = (const float*)d_in[4];
    const float* ln_w  = (const float*)d_in[5];
    const float* ln_b  = (const float*)d_in[6];
    const float* W_out = (const float*)d_in[7];
    const float* b_out = (const float*)d_in[8];
    const unsigned char* mask = (const unsigned char*)d_in[9];
    float* out = (float*)d_out;

    char* ws = (char*)d_ws;
    size_t off = 0;
    float*  fg     = (float*)(ws + off);  off += (size_t)BB * NN * DD * 4;   // 134 MB
    bf16_t* wf_bf  = (bf16_t*)(ws + off); off += (size_t)DD * DD * 2;
    bf16_t* wgm_bf = (bf16_t*)(ws + off); off += (size_t)DD * DD * 2;
    bf16_t* wct    = (bf16_t*)(ws + off); off += (size_t)DD * DD * 2;
    float*  qbuf   = (float*)(ws + off);  off += (size_t)BB * DD * 4;
    float*  bbuf   = (float*)(ws + off);  off += (size_t)BB * NN * 4;
    float*  cbuf   = (float*)(ws + off);  off += (size_t)BB * NN * 4;
    float*  dbuf   = (float*)(ws + off);  off += (size_t)BB * NN * 4;
    float*  oacc   = (float*)(ws + off);  off += (size_t)BB * DD * 4;

    // ---- prep / init ----
    prep_kernel<<<(DD * DD) / 256, 256, 0, stream>>>(W_f, W_g, wf_bf, wgm_bf);
    binit_kernel<<<(BB * NN) / 256, 256, 0, stream>>>(mask, bbuf);
    qinit_kernel<<<dim3(DD / 128, BB), 128, 0, stream>>>(query, W_h, qbuf);

    // ---- combined weight + big GEMM ----
    gemm_wc_kernel<<<dim3(DD / 128, DD / 64), 256, 0, stream>>>(wgm_bf, wf_bf, wct);
    // grid.x = N-tiles (8, fast) ; grid.y = M-tiles (256) -> feat panels L2-hot
    gemm_fg_kernel<<<dim3(DD / 128, (BB * NN) / 128), 256, 0, stream>>>(feat, wct, fg);

    // ---- routing iterations ----
    for (int it = 0; it < NUM_ITERS; ++it) {
        softmax_kernel<<<BB, 256, 0, stream>>>(bbuf, cbuf);
        hipMemsetAsync(oacc, 0, (size_t)BB * DD * 4, stream);
        wsum_kernel<<<dim3(64, BB), 256, 0, stream>>>(cbuf, fg, oacc);
        if (it < NUM_ITERS - 1) {
            delta_kernel<<<(BB * NN) / 8, 256, 0, stream>>>(qbuf, fg, dbuf);
            bupdate_kernel<<<BB, 256, 0, stream>>>(bbuf, dbuf);
        }
        float scale = (it < NUM_ITERS - 1) ? (1.0f / (float)NN) : 1.0f;
        qupdate_kernel<<<BB, 256, 0, stream>>>(qbuf, oacc, ln_w, ln_b, scale);
    }

    // ---- final projection ----
    final_kernel<<<(BB * OUTD) / 8, 256, 0, stream>>>(qbuf, W_out, b_out, out);
    (void)in_sizes; (void)n_in; (void)out_size; (void)ws_size;
}
